// CausalSelfAttention_27874337751659
// MI455X (gfx1250) — compile-verified, matrix operand order checked by reading, not run on hardware
//
#include <hip/hip_runtime.h>
#include <hip/hip_bf16.h>

typedef __attribute__((ext_vector_type(16))) _Float16 v16h;
typedef __attribute__((ext_vector_type(8)))  _Float16 v8h;
typedef __attribute__((ext_vector_type(8)))  float    v8f;

constexpr int Bn = 2, Ln = 2048, Dn = 1024, Hn = 16, HDn = 64;
constexpr float SCALE = 0.125f; // 64^-0.5

__device__ __forceinline__ v8f wmma_f16(v16h a, v16h b, v8f c) {
    return __builtin_amdgcn_wmma_f32_16x16x32_f16(false, a, false, b, (short)0, c, false, false);
}

// A-fragment (16x32 f16 rows) / B-fragment (32x16 columns) for
// v_wmma_f32_16x16x32_f16. Lane holds row/col (lane&15); K runs
// [8h, 8h+8) and [16+8h, 16+8h+8), h = lane>>4.
__device__ __forceinline__ v16h load_frag(const _Float16* __restrict__ base, int ld) {
    const int lane = threadIdx.x & 31;
    const int r    = lane & 15;
    const int kb   = (lane >> 4) << 3;
    const _Float16* p = base + r * ld + kb;
    union { v16h v; v8h h2[2]; } u;
    u.h2[0] = *(const v8h*)(p);
    u.h2[1] = *(const v8h*)(p + 16);
    return u.v;
}

// ---------------- stage 0: precision conversion / weight transpose ----------
__global__ void f32_to_f16_kernel(const float* __restrict__ in, _Float16* __restrict__ out, int n) {
    int i = blockIdx.x * blockDim.x + threadIdx.x;
    if (i < n) out[i] = (_Float16)in[i];
}

// in: R x C fp32; out: C x R f16 (transposed)
__global__ void transpose_f16_kernel(const float* __restrict__ in, _Float16* __restrict__ out,
                                     int R, int C) {
    int i = blockIdx.x * blockDim.x + threadIdx.x;
    if (i < R * C) {
        int r2 = i / R;   // output row = input column
        int c2 = i % R;   // output col = input row
        out[i] = (_Float16)in[c2 * C + r2];
    }
}

// ---------------- stage 1: QKV projection GEMM -------------------------------
// X (4096 x 1024) * WqkvT (3072 x 1024, pre-transposed) -> scatter q/k/v.
// One wave = 32x64 output tile; ping-pong double buffer (no register rotation).
__global__ void __launch_bounds__(256) qkv_gemm_kernel(
    const _Float16* __restrict__ Xh, const _Float16* __restrict__ WqT,
    const float* __restrict__ bias,
    _Float16* __restrict__ qh, _Float16* __restrict__ kh, _Float16* __restrict__ vt,
    float* __restrict__ kout, float* __restrict__ vout)
{
    const int lane = threadIdx.x & 31;
    const int wid  = blockIdx.x * 8 + (threadIdx.x >> 5); // 0..6143
    const int mt   = wid / 48;   // M tile (32 rows), 0..127
    const int ns   = wid % 48;   // N strip (64 cols), 0..47

    const _Float16* A0 = Xh + (size_t)mt * 32 * Dn;
    const _Float16* A1 = A0 + 16 * Dn;
    const _Float16* Bb = WqT + (size_t)ns * 64 * Dn;

    v8f acc[2][4] = {};
    v16h a0[2], b0[4], a1[2], b1[4];
    a0[0] = load_frag(A0, Dn);
    a0[1] = load_frag(A1, Dn);
#pragma unroll
    for (int t = 0; t < 4; ++t) b0[t] = load_frag(Bb + t * 16 * Dn, Dn);

    for (int k0 = 0; k0 < Dn; k0 += 64) {
        // prefetch odd k-step (k0+32 < Dn always: Dn % 64 == 0)
        a1[0] = load_frag(A0 + k0 + 32, Dn);
        a1[1] = load_frag(A1 + k0 + 32, Dn);
#pragma unroll
        for (int t = 0; t < 4; ++t) b1[t] = load_frag(Bb + t * 16 * Dn + k0 + 32, Dn);

#pragma unroll
        for (int mi = 0; mi < 2; ++mi)
#pragma unroll
            for (int t = 0; t < 4; ++t)
                acc[mi][t] = wmma_f16(a0[mi], b0[t], acc[mi][t]);

        if (k0 + 64 < Dn) {   // prefetch next even k-step
            a0[0] = load_frag(A0 + k0 + 64, Dn);
            a0[1] = load_frag(A1 + k0 + 64, Dn);
#pragma unroll
            for (int t = 0; t < 4; ++t) b0[t] = load_frag(Bb + t * 16 * Dn + k0 + 64, Dn);
        }

#pragma unroll
        for (int mi = 0; mi < 2; ++mi)
#pragma unroll
            for (int t = 0; t < 4; ++t)
                acc[mi][t] = wmma_f16(a1[mi], b1[t], acc[mi][t]);
    }

    const int part = ns >> 4;        // 0=q 1=k 2=v (wave-uniform)
    const int h    = ns & 15;        // head (wave-uniform)
    const int hs   = lane >> 4;
    const int nl   = lane & 15;
#pragma unroll
    for (int mi = 0; mi < 2; ++mi) {
#pragma unroll
        for (int t = 0; t < 4; ++t) {
#pragma unroll
            for (int r = 0; r < 8; ++r) {
                int row = mt * 32 + mi * 16 + r + 8 * hs;   // 0..4095
                int col = ns * 64 + t * 16 + nl;            // 0..3071
                float val = acc[mi][t][r] + bias[col];
                int bb = row >> 11, l = row & 2047;
                int hd = t * 16 + nl;
                int idx = ((bb * Hn + h) * Ln + l) * HDn + hd;
                if (part == 0) {
                    qh[idx] = (_Float16)val;
                } else if (part == 1) {
                    kout[idx] = val;
                    kh[idx]   = (_Float16)val;
                } else {
                    vout[idx] = val;
                    vt[((bb * Hn + h) * HDn + hd) * Ln + l] = (_Float16)val; // transposed
                }
            }
        }
    }
}

// ---------------- stage 2: flash attention ----------------------------------
// one wave = 16 query rows of one (b,h); online softmax over 32-key chunks
__global__ void __launch_bounds__(256) attn_kernel(
    const _Float16* __restrict__ qh, const _Float16* __restrict__ kh,
    const _Float16* __restrict__ vt, _Float16* __restrict__ aoh)
{
    __shared__ __align__(16) _Float16 pbuf[8 * 512];
    const int lane = threadIdx.x & 31;
    const int widx = threadIdx.x >> 5;
    const int w    = blockIdx.x * 8 + widx;  // 0..4095
    const int bh   = w >> 7;                 // (b*16+h)
    const int qt   = w & 127;
    const int qbase = qt * 16;

    const _Float16* Qp = qh + (size_t)bh * Ln * HDn;
    const _Float16* Kp = kh + (size_t)bh * Ln * HDn;
    const _Float16* Vp = vt + (size_t)bh * HDn * Ln;
    _Float16* pb = &pbuf[widx * 512];

    v16h aq0 = load_frag(Qp + qbase * HDn, HDn);
    v16h aq1 = load_frag(Qp + qbase * HDn + 32, HDn);

    v8f o0 = {}, o1 = {}, o2 = {}, o3 = {};
    float mi[8], li[8];
#pragma unroll
    for (int r = 0; r < 8; ++r) { mi[r] = -3.0e38f; li[r] = 0.f; }

    const int hs = lane >> 4;
    const int nl = lane & 15;
    const int nchunks = qt / 2 + 1;          // causal: keys <= qbase+15

    for (int c = 0; c < nchunks; ++c) {
        const int sbase = c * 32;

        // issue all K and V fragment loads up front; V latency hides under
        // the QK WMMAs + softmax VALU work
        v16h bk0 = load_frag(Kp + (size_t)sbase * HDn,             HDn);
        v16h bk1 = load_frag(Kp + (size_t)sbase * HDn + 32,        HDn);
        v16h bk2 = load_frag(Kp + (size_t)(sbase + 16) * HDn,      HDn);
        v16h bk3 = load_frag(Kp + (size_t)(sbase + 16) * HDn + 32, HDn);
        v16h bv0 = load_frag(Vp + 0 * 16 * Ln + sbase, Ln);
        v16h bv1 = load_frag(Vp + 1 * 16 * Ln + sbase, Ln);
        v16h bv2 = load_frag(Vp + 2 * 16 * Ln + sbase, Ln);
        v16h bv3 = load_frag(Vp + 3 * 16 * Ln + sbase, Ln);

        v8f s0 = {}, s1 = {};
        s0 = wmma_f16(aq0, bk0, s0);
        s0 = wmma_f16(aq1, bk1, s0);
        s1 = wmma_f16(aq0, bk2, s1);
        s1 = wmma_f16(aq1, bk3, s1);

        float mx[8];
#pragma unroll
        for (int r = 0; r < 8; ++r) {
            int row = qbase + r + 8 * hs;
            int c0  = sbase + nl;
            float v0 = (c0      <= row) ? s0[r] * SCALE : -3.0e38f;
            float v1 = (c0 + 16 <= row) ? s1[r] * SCALE : -3.0e38f;
            s0[r] = v0; s1[r] = v1;
            mx[r] = fmaxf(v0, v1);
        }
#pragma unroll
        for (int off = 1; off < 16; off <<= 1)
#pragma unroll
            for (int r = 0; r < 8; ++r)
                mx[r] = fmaxf(mx[r], __shfl_xor(mx[r], off, 32));

        float corr[8];
#pragma unroll
        for (int r = 0; r < 8; ++r) {
            float mnew = fmaxf(mi[r], mx[r]);
            corr[r] = __expf(mi[r] - mnew);
            mi[r] = mnew;
            float p0 = __expf(s0[r] - mnew);
            float p1 = __expf(s1[r] - mnew);
            s0[r] = p0; s1[r] = p1;
            li[r] *= corr[r];
            mx[r] = p0 + p1;                  // reuse as row-sum accumulator
        }
#pragma unroll
        for (int off = 1; off < 16; off <<= 1)
#pragma unroll
            for (int r = 0; r < 8; ++r)
                mx[r] += __shfl_xor(mx[r], off, 32);
#pragma unroll
        for (int r = 0; r < 8; ++r) {
            li[r] += mx[r];
            o0[r] *= corr[r]; o1[r] *= corr[r]; o2[r] *= corr[r]; o3[r] *= corr[r];
        }

        // C-tile -> A-fragment relayout of P (16x32 f16) via LDS
#pragma unroll
        for (int r = 0; r < 8; ++r) {
            int row = r + 8 * hs;
            pb[row * 32 + nl]      = (_Float16)s0[r];
            pb[row * 32 + 16 + nl] = (_Float16)s1[r];
        }
        const int kb = hs << 3;
        union { v16h v; v8h h2[2]; } up;
        up.h2[0] = *(const v8h*)(pb + nl * 32 + kb);
        up.h2[1] = *(const v8h*)(pb + nl * 32 + kb + 16);
        v16h ap = up.v;

        o0 = wmma_f16(ap, bv0, o0);
        o1 = wmma_f16(ap, bv1, o1);
        o2 = wmma_f16(ap, bv2, o2);
        o3 = wmma_f16(ap, bv3, o3);
    }

    const int bb = bh >> 4, h = bh & 15;
#pragma unroll
    for (int r = 0; r < 8; ++r) {
        float inv = 1.0f / li[r];
        int row = qbase + r + 8 * hs;
        _Float16* dst = aoh + (size_t)(bb * Ln + row) * Dn + h * HDn;
        dst[ 0 + nl] = (_Float16)(o0[r] * inv);
        dst[16 + nl] = (_Float16)(o1[r] * inv);
        dst[32 + nl] = (_Float16)(o2[r] * inv);
        dst[48 + nl] = (_Float16)(o3[r] * inv);
    }
}

// ---------------- stage 3: output projection GEMM ---------------------------
// One wave = 32x64 output tile; ping-pong double buffer (no register rotation).
__global__ void __launch_bounds__(256) out_gemm_kernel(
    const _Float16* __restrict__ Ah, const _Float16* __restrict__ WoT,
    const float* __restrict__ bias, float* __restrict__ out)
{
    const int lane = threadIdx.x & 31;
    const int wid  = blockIdx.x * 8 + (threadIdx.x >> 5); // 0..2047
    const int mt   = wid >> 4;   // 0..127 (32 rows)
    const int ns   = wid & 15;   // 0..15  (64 cols)

    const _Float16* A0 = Ah + (size_t)mt * 32 * Dn;
    const _Float16* A1 = A0 + 16 * Dn;
    const _Float16* Bb = WoT + (size_t)ns * 64 * Dn;

    v8f acc[2][4] = {};
    v16h a0[2], b0[4], a1[2], b1[4];
    a0[0] = load_frag(A0, Dn);
    a0[1] = load_frag(A1, Dn);
#pragma unroll
    for (int t = 0; t < 4; ++t) b0[t] = load_frag(Bb + t * 16 * Dn, Dn);

    for (int k0 = 0; k0 < Dn; k0 += 64) {
        a1[0] = load_frag(A0 + k0 + 32, Dn);
        a1[1] = load_frag(A1 + k0 + 32, Dn);
#pragma unroll
        for (int t = 0; t < 4; ++t) b1[t] = load_frag(Bb + t * 16 * Dn + k0 + 32, Dn);

#pragma unroll
        for (int mi = 0; mi < 2; ++mi)
#pragma unroll
            for (int t = 0; t < 4; ++t)
                acc[mi][t] = wmma_f16(a0[mi], b0[t], acc[mi][t]);

        if (k0 + 64 < Dn) {
            a0[0] = load_frag(A0 + k0 + 64, Dn);
            a0[1] = load_frag(A1 + k0 + 64, Dn);
#pragma unroll
            for (int t = 0; t < 4; ++t) b0[t] = load_frag(Bb + t * 16 * Dn + k0 + 64, Dn);
        }

#pragma unroll
        for (int mi = 0; mi < 2; ++mi)
#pragma unroll
            for (int t = 0; t < 4; ++t)
                acc[mi][t] = wmma_f16(a1[mi], b1[t], acc[mi][t]);
    }

    const int hs = lane >> 4, nl = lane & 15;
#pragma unroll
    for (int mi = 0; mi < 2; ++mi)
#pragma unroll
        for (int t = 0; t < 4; ++t)
#pragma unroll
            for (int r = 0; r < 8; ++r) {
                int row = mt * 32 + mi * 16 + r + 8 * hs;
                int col = ns * 64 + t * 16 + nl;
                out[(size_t)row * Dn + col] = acc[mi][t][r] + bias[col];
            }
}

extern "C" void kernel_launch(void* const* d_in, const int* in_sizes, int n_in,
                              void* d_out, int out_size, void* d_ws, size_t ws_size,
                              hipStream_t stream) {
    const float* x     = (const float*)d_in[0];
    const float* w_qkv = (const float*)d_in[1];
    const float* b_qkv = (const float*)d_in[2];
    const float* w_out = (const float*)d_in[3];
    const float* b_out = (const float*)d_in[4];

    float* out  = (float*)d_out;                    // (B,L,D)
    float* kout = out  + (size_t)Bn * Ln * Dn;      // (B,H,L,HD)
    float* vout = kout + (size_t)Bn * Ln * Dn;      // (B,H,L,HD)

    _Float16* xh  = (_Float16*)d_ws;                        // B*L*D
    _Float16* wqT = xh  + (size_t)Bn * Ln * Dn;             // 3D x D (transposed)
    _Float16* woT = wqT + (size_t)3 * Dn * Dn;              // D x D (transposed)
    _Float16* qh  = woT + (size_t)Dn * Dn;                  // (B,H,L,HD)
    _Float16* kh  = qh  + (size_t)Bn * Ln * Dn;             // (B,H,L,HD)
    _Float16* vt  = kh  + (size_t)Bn * Ln * Dn;             // (B,H,HD,L)
    _Float16* aoh = vt  + (size_t)Bn * Ln * Dn;             // (B*L, D)

    const int nx = Bn * Ln * Dn;
    f32_to_f16_kernel<<<(nx + 255) / 256, 256, 0, stream>>>(x, xh, nx);
    transpose_f16_kernel<<<(3 * Dn * Dn + 255) / 256, 256, 0, stream>>>(w_qkv, wqT, Dn, 3 * Dn);
    transpose_f16_kernel<<<(Dn * Dn + 255) / 256, 256, 0, stream>>>(w_out, woT, Dn, Dn);

    qkv_gemm_kernel<<<768, 256, 0, stream>>>(xh, wqT, b_qkv, qh, kh, vt, kout, vout);
    attn_kernel<<<512, 256, 0, stream>>>(qh, kh, vt, aoh);
    out_gemm_kernel<<<256, 256, 0, stream>>>(aoh, woT, b_out, out);
}
